// GLA_GCN_86715389706285
// MI455X (gfx1250) — matrix operand order repaired and checked by reading, not compiled, not run on hardware
//
#include <hip/hip_runtime.h>
#include <hip/hip_bf16.h>

// ---------------------------------------------------------------------------
// GLA-GCN on MI455X (gfx1250): bf16 WMMA GEMMs + fused adjacency-mix epilogues.
// Partition 1 of the local branch is dead (all joints overwritten) -> skipped.
// GEMM: 256x64 block tile, 8 waves x (32Mx64N), double-buffered LDS,
//       weights pre-transposed so B staging is a single ds_store_b128.
// ---------------------------------------------------------------------------

typedef unsigned int   u32;
typedef unsigned short u16;
typedef __attribute__((ext_vector_type(4)))  u32    u32x4;
typedef __attribute__((ext_vector_type(16))) __bf16 v16bf;
typedef __attribute__((ext_vector_type(8)))  float  v8f;

#define B_DIM   4096
#define J_DIM   25
#define NROW    (B_DIM * J_DIM)   // 102400
#define D_IN    512
#define D_H     256
#define D_OUT   512

__constant__ int c_pidx[5][6] = {
    {0, 1, 2, 3, 4, 0},
    {5, 6, 11, 12, 0, 0},
    {5, 7, 9, 0, 0, 0},
    {6, 8, 10, 0, 0, 0},
    {11, 12, 13, 14, 15, 16}};
__constant__ int c_adj_n[5]   = {25, 5, 3, 3, 6};        // global, p0, p2, p3, p4
__constant__ int c_adj_off[5] = {0, 640, 704, 768, 832}; // float offsets in ADJ pool

static __device__ __forceinline__ u16 f2bf(float f) {
    u32 u = __float_as_uint(f);
    u += 0x7FFFu + ((u >> 16) & 1u);   // round-to-nearest-even
    return (u16)(u >> 16);
}

// ---------------------------------------------------------------------------
// Adjacency normalization: norm = d^-1/2 * A * d^-1/2, d = clip(rowsum, 1e-6).
// ---------------------------------------------------------------------------
__global__ void adjnorm_kernel(const float* __restrict__ a0, const float* __restrict__ a1,
                               const float* __restrict__ a2, const float* __restrict__ a3,
                               const float* __restrict__ a4, float* __restrict__ out) {
    __shared__ float d[32];
    const int m = blockIdx.x;
    const float* A = (m == 0) ? a0 : (m == 1) ? a1 : (m == 2) ? a2 : (m == 3) ? a3 : a4;
    const int n = c_adj_n[m];
    float* o = out + c_adj_off[m];
    const int t = threadIdx.x;
    if (t < n) {
        float s = 0.f;
        for (int j = 0; j < n; ++j) s += A[t * n + j];
        d[t] = rsqrtf(fmaxf(s, 1e-6f));
    }
    __syncthreads();
    for (int e = t; e < n * n; e += blockDim.x) {
        int i = e / n, j = e % n;
        o[e] = d[i] * A[e] * d[j];
    }
}

// f32 -> bf16 (grid-strided), row-major preserved.
__global__ void cvt_bf16_kernel(const float* __restrict__ s, u16* __restrict__ d, size_t n) {
    size_t i  = (size_t)blockIdx.x * blockDim.x + threadIdx.x;
    size_t st = (size_t)gridDim.x * blockDim.x;
    for (; i < n; i += st) d[i] = f2bf(s[i]);
}

// f32 (KxN row-major) -> bf16 transposed (NxK row-major). dst writes coalesced.
__global__ void cvtT_bf16_kernel(const float* __restrict__ s, u16* __restrict__ d,
                                 int Kd, int Nd) {
    size_t gid = (size_t)blockIdx.x * blockDim.x + threadIdx.x;
    if (gid >= (size_t)Kd * Nd) return;
    int k = (int)(gid % Kd);
    int n = (int)(gid / Kd);
    d[(size_t)n * Kd + k] = f2bf(s[(size_t)k * Nd + n]);
}

__global__ void zero_kernel(u32x4* __restrict__ d, size_t n4) {
    size_t i  = (size_t)blockIdx.x * blockDim.x + threadIdx.x;
    size_t st = (size_t)gridDim.x * blockDim.x;
    u32x4 z = {0u, 0u, 0u, 0u};
    for (; i < n4; i += st) d[i] = z;
}

// Gather partition joints from x_bf16 into compact (B*np, 512) bf16 rows.
__global__ void gather_kernel(const u16* __restrict__ src, u16* __restrict__ dst,
                              int np, int part) {
    const size_t total = (size_t)B_DIM * np * (D_IN / 8);
    size_t gid = (size_t)blockIdx.x * blockDim.x + threadIdx.x;
    if (gid >= total) return;
    int v = (int)(gid % (D_IN / 8));
    int i = (int)((gid / (D_IN / 8)) % np);
    int b = (int)(gid / ((size_t)(D_IN / 8) * np));
    int srow = b * J_DIM + c_pidx[part][i];
    *(u32x4*)(dst + ((size_t)(b * np + i)) * D_IN + v * 8) =
        *(const u32x4*)(src + (size_t)srow * D_IN + v * 8);
}

// Adjacency mix + bias + relu + bf16 store with optional joint scatter.
__global__ void adjmix_kernel(const float* __restrict__ H, const float* __restrict__ A,
                              const float* __restrict__ bias, u16* __restrict__ out,
                              int nj, int F, int ldo, int ccol0, int oj, int part) {
    const size_t total = (size_t)B_DIM * nj * F;
    size_t gid = (size_t)blockIdx.x * blockDim.x + threadIdx.x;
    if (gid >= total) return;
    int f = (int)(gid % F);
    int i = (int)((gid / F) % nj);
    int b = (int)(gid / ((size_t)F * nj));
    float acc = bias[f];
    const float* Ai = A + i * nj;
    const float* Hb = H + (size_t)b * nj * F + f;
    for (int j = 0; j < nj; ++j) acc += Ai[j] * Hb[(size_t)j * F];
    int joint = (part < 0) ? i : c_pidx[part][i];
    size_t orow = (size_t)b * oj + joint;
    out[orow * ldo + ccol0 + f] = f2bf(fmaxf(acc, 0.f));
}

// ---------------------------------------------------------------------------
// bf16 WMMA GEMM: C(MxN) = A(MxK) * Bt^T, where Bt is the N x K transposed
// weight (row-major). fp32 accumulate. Block tile 256x64, K-step 32,
// 256 threads = 8 waves; wave tile 32x64 = 8 WMMAs (B frags reused).
// Double-buffered LDS. Requires M%256==0, N%64==0, K%32==0.
// MODE 0: raw f32 out; 1: bias+relu -> bf16; 2: bias -> f32 (nontemporal).
// ---------------------------------------------------------------------------
#define BM 256
#define BN 64
#define BK 32
#define PA 40   // LDS row pitch (u16): 32 + 8 pad (80B, 16B aligned)
#define PB 40

template <int MODE>
__global__ __launch_bounds__(256)
void gemm_bf16_kernel(const u16* __restrict__ A, int lda,
                      const u16* __restrict__ Bt, int ldb /* = K */, int K,
                      void* __restrict__ Cout, int ldc,
                      const float* __restrict__ bias) {
    __shared__ u16 As[2][BM * PA];   // 256 x 32 tile
    __shared__ u16 Bs[2][BN * PB];   // Bs[n][k], 64 x 32 tile

    const int tid  = threadIdx.x;
    const int wave = tid >> 5;
    const int lane = tid & 31;
    const int ln   = lane & 15;
    const int lh   = lane >> 4;
    const int m0   = blockIdx.x * BM;
    const int n0   = blockIdx.y * BN;

    // Staging assignment: A -> one full row (32 elems, 64B) per thread;
    //                     B -> 8 elems per thread from transposed weight.
    const int brow = tid >> 2, bcol = (tid & 3) * 8;
    const u16* Aball = A + (size_t)(m0 + tid) * lda;
    const u16* Bball = Bt + (size_t)(n0 + brow) * ldb + bcol;

    v8f acc[2][4];
#pragma unroll
    for (int s = 0; s < 2; ++s)
#pragma unroll
        for (int j = 0; j < 4; ++j) acc[s][j] = v8f{};

    u32x4 ar[4], br;
    // Prologue: fetch tile 0.
    {
        const u32x4* pa = (const u32x4*)(Aball);
#pragma unroll
        for (int q = 0; q < 4; ++q) ar[q] = pa[q];
        br = *(const u32x4*)(Bball);
        u32x4* da = (u32x4*)(&As[0][tid * PA]);
#pragma unroll
        for (int q = 0; q < 4; ++q) da[q] = ar[q];
        *(u32x4*)(&Bs[0][brow * PB + bcol]) = br;
    }
    __syncthreads();

    int buf = 0;
    for (int k0 = 0; k0 < K; k0 += BK) {
        const bool has_next = (k0 + BK) < K;
        if (has_next) {
            const u32x4* pa = (const u32x4*)(Aball + k0 + BK);
#pragma unroll
            for (int q = 0; q < 4; ++q) ar[q] = pa[q];
            br = *(const u32x4*)(Bball + k0 + BK);
        }
        if (k0 + 2 * BK < K) {
            __builtin_prefetch(Aball + k0 + 2 * BK, 0, 1);
            __builtin_prefetch(Bball + k0 + 2 * BK, 0, 1);
        }

        // ---- compute from As[buf]/Bs[buf]: issue ALL frag loads first so the
        // scheduler can stagger waits and run the 8 WMMAs back-to-back. ----
        union Frag { u32x4 q[2]; v16bf v; };
        Frag fa[2], fb[4];
#pragma unroll
        for (int s = 0; s < 2; ++s) {
            const u16* base = &As[buf][(wave * 32 + s * 16 + ln) * PA];
            fa[s].q[0] = *(const u32x4*)(base + 8 * lh);
            fa[s].q[1] = *(const u32x4*)(base + 16 + 8 * lh);
        }
#pragma unroll
        for (int j = 0; j < 4; ++j) {
            const u16* nb = &Bs[buf][(j * 16 + ln) * PB];
            fb[j].q[0] = *(const u32x4*)(nb + 8 * lh);
            fb[j].q[1] = *(const u32x4*)(nb + 16 + 8 * lh);
        }
#pragma unroll
        for (int s = 0; s < 2; ++s)
#pragma unroll
            for (int j = 0; j < 4; ++j)
                acc[s][j] = __builtin_amdgcn_wmma_f32_16x16x32_bf16(
                    false, fa[s].v, false, fb[j].v, (short)0, acc[s][j], false, false);

        if (has_next) {
            u32x4* da = (u32x4*)(&As[buf ^ 1][tid * PA]);
#pragma unroll
            for (int q = 0; q < 4; ++q) da[q] = ar[q];
            *(u32x4*)(&Bs[buf ^ 1][brow * PB + bcol]) = br;
        }
        __syncthreads();
        buf ^= 1;
    }

    // Epilogue (C/D layout: VGPR r -> M = r + 8*lh, N = ln).
    // Per-tile base pointer + ldc bump: one 64-bit add per store, no muls.
#pragma unroll
    for (int s = 0; s < 2; ++s) {
        const int crow = m0 + wave * 32 + s * 16 + 8 * lh;
#pragma unroll
        for (int j = 0; j < 4; ++j) {
            const int cn = n0 + j * 16 + ln;
            if (MODE == 0) {
                float* p = (float*)Cout + (size_t)crow * ldc + cn;
#pragma unroll
                for (int r = 0; r < 8; ++r) { *p = acc[s][j][r]; p += ldc; }
            } else if (MODE == 1) {
                u16* p = (u16*)Cout + (size_t)crow * ldc + cn;
                const float bv = bias[cn];
#pragma unroll
                for (int r = 0; r < 8; ++r) {
                    *p = f2bf(fmaxf(acc[s][j][r] + bv, 0.f));
                    p += ldc;
                }
            } else {
                float* p = (float*)Cout + (size_t)crow * ldc + cn;
                const float bv = bias[cn];
#pragma unroll
                for (int r = 0; r < 8; ++r) {
                    __builtin_nontemporal_store(acc[s][j][r] + bv, p);
                    p += ldc;
                }
            }
        }
    }
}

// ---------------------------------------------------------------------------
// Host orchestration.
// ---------------------------------------------------------------------------
static inline void cvt(const void* src, u16* dst, size_t n, hipStream_t s) {
    cvt_bf16_kernel<<<2048, 256, 0, s>>>((const float*)src, dst, n);
}
static inline void cvtT(const void* src, u16* dst, int K, int N, hipStream_t s) {
    size_t tot = (size_t)K * N;
    cvtT_bf16_kernel<<<(tot + 255) / 256, 256, 0, s>>>((const float*)src, dst, K, N);
}

extern "C" void kernel_launch(void* const* d_in, const int* in_sizes, int n_in,
                              void* d_out, int out_size, void* d_ws, size_t ws_size,
                              hipStream_t stream) {
    (void)in_sizes; (void)n_in; (void)out_size; (void)ws_size;

    char* ws = (char*)d_ws;
    const size_t oXBF = 0;                                     // x bf16: 102400x512
    const size_t oXC  = oXBF + (size_t)NROW * D_IN * 2;        // concat bf16: 102400x1024
    const size_t oH   = oXC + (size_t)NROW * 1024 * 2;         // f32 scratch: 102400x512
    const size_t oT   = oH + (size_t)NROW * 512 * 4;           // bf16 scratch: 102400x512
    const size_t oW   = oT + (size_t)NROW * 512 * 2;           // bf16 weight pool (transposed)
    const size_t oADJ = oW + 2097152ull * 2;                   // normalized adjacencies

    u16*   XBF = (u16*)(ws + oXBF);
    u16*   XC  = (u16*)(ws + oXC);
    float* H   = (float*)(ws + oH);
    u16*   T   = (u16*)(ws + oT);
    u16*   WP  = (u16*)(ws + oW);
    float* ADJ = (float*)(ws + oADJ);
    u16*   G2  = (u16*)(ws + oT + (64ull << 20));              // local mid activations

    // bf16 transposed weight pool offsets (elements).
    u16* wg0 = WP;                         // (256x512)  Wg0^T
    u16* wg1 = WP + 131072;                // (512x256)  Wg1^T
    u16* wl  = WP + 262144;                // 4 parts x (Wla^T + Wlb^T)
    u16* wf0 = WP + 262144 + 4 * 262144;   // (512x1024) Wf0^T
    u16* wf1 = wf0 + 524288;               // (512x512)  Wf1^T

    // 1. Normalized adjacencies (global + parts 0,2,3,4).
    adjnorm_kernel<<<5, 64, 0, stream>>>(
        (const float*)d_in[1], (const float*)d_in[2], (const float*)d_in[4],
        (const float*)d_in[5], (const float*)d_in[6], ADJ);

    // 2. Convert x (row-major) and live weights (transposed) to bf16; zero concat.
    cvt(d_in[0], XBF, (size_t)NROW * D_IN, stream);
    cvtT(d_in[7], wg0, D_IN, D_H, stream);
    cvtT(d_in[9], wg1, D_H, D_OUT, stream);
    const int parts[4] = {0, 2, 3, 4};
    const int pn[4]    = {5, 3, 3, 6};
    for (int s = 0; s < 4; ++s) {
        int p = parts[s];
        cvtT(d_in[11 + 4 * p], wl + s * 262144,          D_IN, D_H, stream);
        cvtT(d_in[13 + 4 * p], wl + s * 262144 + 131072, D_H, D_OUT, stream);
    }
    cvtT(d_in[31], wf0, 1024, D_OUT, stream);
    cvtT(d_in[33], wf1, D_OUT, D_OUT, stream);
    zero_kernel<<<2048, 256, 0, stream>>>((u32x4*)XC, (size_t)NROW * 1024 / 8);

    const int TPB = 256;

    // 3. Global branch.
    gemm_bf16_kernel<0><<<dim3(NROW / BM, D_H / BN), TPB, 0, stream>>>(
        XBF, D_IN, wg0, D_IN, D_IN, H, D_H, nullptr);
    {
        size_t tot = (size_t)NROW * D_H;
        adjmix_kernel<<<(tot + TPB - 1) / TPB, TPB, 0, stream>>>(
            H, ADJ, (const float*)d_in[8], T, J_DIM, D_H, D_H, 0, J_DIM, -1);
    }
    gemm_bf16_kernel<0><<<dim3(NROW / BM, D_OUT / BN), TPB, 0, stream>>>(
        T, D_H, wg1, D_H, D_H, H, D_OUT, nullptr);
    {
        size_t tot = (size_t)NROW * D_OUT;
        adjmix_kernel<<<(tot + TPB - 1) / TPB, TPB, 0, stream>>>(
            H, ADJ, (const float*)d_in[10], XC, J_DIM, D_OUT, 1024, 0, J_DIM, -1);
    }

    // 4. Local branches (parts 0,2,3,4 only — part 1 fully overwritten).
    for (int s = 0; s < 4; ++s) {
        const int p = parts[s], np = pn[s];
        const int rows = B_DIM * np;                  // 20480/12288/12288/24576: %256==0
        const float* adjp = ADJ + 640 + 64 * s;
        u16* wla = wl + s * 262144;
        u16* wlb = wla + 131072;
        {
            size_t tot = (size_t)rows * (D_IN / 8);
            gather_kernel<<<(tot + TPB - 1) / TPB, TPB, 0, stream>>>(XBF, T, np, p);
        }
        gemm_bf16_kernel<0><<<dim3(rows / BM, D_H / BN), TPB, 0, stream>>>(
            T, D_IN, wla, D_IN, D_IN, H, D_H, nullptr);
        {
            size_t tot = (size_t)rows * D_H;
            adjmix_kernel<<<(tot + TPB - 1) / TPB, TPB, 0, stream>>>(
                H, adjp, (const float*)d_in[12 + 4 * p], G2, np, D_H, D_H, 0, np, -1);
        }
        gemm_bf16_kernel<0><<<dim3(rows / BM, D_OUT / BN), TPB, 0, stream>>>(
            G2, D_H, wlb, D_H, D_H, H, D_OUT, nullptr);
        {
            size_t tot = (size_t)rows * D_OUT;
            adjmix_kernel<<<(tot + TPB - 1) / TPB, TPB, 0, stream>>>(
                H, adjp, (const float*)d_in[14 + 4 * p], XC, np, D_OUT, 1024, 512, J_DIM, p);
        }
    }

    // 5. Fusion: T = relu(XC @ Wf0 + bf0) bf16; d_out = T @ Wf1 + bf1 (f32, NT).
    gemm_bf16_kernel<1><<<dim3(NROW / BM, D_OUT / BN), TPB, 0, stream>>>(
        XC, 1024, wf0, 1024, 1024, T, D_OUT, (const float*)d_in[32]);
    gemm_bf16_kernel<2><<<dim3(NROW / BM, D_OUT / BN), TPB, 0, stream>>>(
        T, D_OUT, wf1, D_OUT, D_OUT, d_out, D_OUT, (const float*)d_in[34]);
}
